// DeepSDE_82592221102257
// MI455X (gfx1250) — matrix operand order repaired and checked by reading, not compiled, not run on hardware
//
#include <hip/hip_runtime.h>
#include <hip/hip_bf16.h>
#include <cmath>

typedef __attribute__((ext_vector_type(16))) _Float16 v16h;
typedef __attribute__((ext_vector_type(8)))  float    v8f;

// gfx1250 has a hardware v_tanh_f32 (TRANS pipe) — the kernel is tanh-bound,
// so use it if the toolchain exposes it; fall back to libm otherwise.
#if __has_builtin(__builtin_amdgcn_tanhf)
#define TANHF(x) __builtin_amdgcn_tanhf(x)
#elif __has_builtin(__builtin_amdgcn_tanh_f32)
#define TANHF(x) __builtin_amdgcn_tanh_f32(x)
#else
#define TANHF(x) tanhf(x)
#endif

// One wave = 16 SDE paths, replicated across the two lane halves.
//   lane l: m = l&15 (path), hi = l>>4 (which K half it produces for B)
// Per step, per MLP:
//   lanes compute h[k]=tanh(fma(y,w1[k],b1[k])) for their 2x16 K-slots (f32),
//   convert to f16 in the exact B(32x16) WMMA layout, and two chained
//   v_wmma_f32_16x16x32_f16 with A = w2 broadcast into all 16 rows produce
//   D[0] = dot(h, w2) identically in every lane -> no shuffles back.
__global__ __launch_bounds__(256)
void deepsde_em_kernel(const float* __restrict__ y0g,
                       const float* __restrict__ times,
                       const float* __restrict__ noise,
                       const float* __restrict__ f1w, const float* __restrict__ f1b,
                       const float* __restrict__ f2w, const float* __restrict__ f2b,
                       const float* __restrict__ g1w, const float* __restrict__ g1b,
                       const float* __restrict__ g2w, const float* __restrict__ g2b,
                       float* __restrict__ out, int N, int T)
{
    __shared__ float dt2[600][2];     // (dt, sqrt(dt)) per step
    __shared__ float nbuf[8][2][16];  // per-wave double-buffered noise staging

    const int tid  = threadIdx.x;
    const int lane = tid & 31;
    const int w    = tid >> 5;        // wave in block (8 waves)
    const int m    = lane & 15;       // path within wave group
    const int hi   = (lane >> 4) & 1; // lane half -> K half
    const int Tm1  = T - 1;

    for (int i = tid; i < Tm1; i += blockDim.x) {
        float dt  = times[i + 1] - times[i];
        dt2[i][0] = dt;
        dt2[i][1] = __builtin_sqrtf(dt);
    }
    __syncthreads();

    const int path = blockIdx.x * 128 + w * 16 + m;

    // ---- per-lane layer-1 weights for this lane's K slots (kept in VGPRs) ----
    float w1f[32], b1f[32], w1g[32], b1g[32];
#pragma unroll
    for (int c = 0; c < 2; ++c)
#pragma unroll
        for (int j = 0; j < 16; ++j) {
            const int k = 32 * c + (hi ? 16 : 0) + j;
            const int i = c * 16 + j;
            w1f[i] = f1w[k]; b1f[i] = f1b[k];
            w1g[i] = g1w[k]; b1g[i] = g1b[k];
        }

    // ---- A operand (loop-invariant): w2 broadcast into all 16 rows ----
    // A(16x32 f16) k order chosen here (contiguous): VGPR r: lanes0-15 K=2r,2r+1; lanes16-31 +16 (matches B)
    v16h aF[2], aG[2];
#pragma unroll
    for (int c = 0; c < 2; ++c)
#pragma unroll
        for (int r = 0; r < 8; ++r) {
            const int ka = 32 * c + (hi ? 16 : 0) + 2 * r;
            aF[c][2 * r]     = (_Float16)f2w[ka];
            aF[c][2 * r + 1] = (_Float16)f2w[ka + 1];
            aG[c][2 * r]     = (_Float16)g2w[ka];
            aG[c][2 * r + 1] = (_Float16)g2w[ka + 1];
        }

    const float b2f = f2b[0], b2g = g2b[0];

    float y = y0g[path];
    const float yinit = y;

    // LDS byte offsets (generic LDS address low 32 bits == LDS offset)
    const unsigned ldsoff0 = (unsigned)(size_t)(&nbuf[w][0][m]);
    const unsigned ldsoff1 = (unsigned)(size_t)(&nbuf[w][1][m]);

    // prime the async pipeline with step-0 noise
    {
        const unsigned long long g = (unsigned long long)(size_t)(noise + (size_t)0 * N + path);
        asm volatile("global_load_async_to_lds_b32 %0, %1, off"
                     :: "v"(ldsoff0), "v"(g) : "memory");
    }

    for (int t = 0; t < Tm1; ++t) {
        const int cur = t & 1;
        if (t + 1 < Tm1) {
            const unsigned long long g =
                (unsigned long long)(size_t)(noise + (size_t)(t + 1) * N + path);
            const unsigned loff = cur ? ldsoff0 : ldsoff1;
            asm volatile("global_load_async_to_lds_b32 %0, %1, off"
                         :: "v"(loff), "v"(g) : "memory");
            asm volatile("s_wait_asynccnt 0x1" ::: "memory"); // step-t staging done
        } else {
            asm volatile("s_wait_asynccnt 0x0" ::: "memory");
        }

        v8f df = {}; v8f dg = {};
#pragma unroll
        for (int c = 0; c < 2; ++c) {
            v16h hf, hg;
#pragma unroll
            for (int r = 0; r < 8; ++r) {
                const int i0 = c * 16 + 2 * r;
                hf[2 * r]     = (_Float16)TANHF(__builtin_fmaf(y, w1f[i0],     b1f[i0]));
                hf[2 * r + 1] = (_Float16)TANHF(__builtin_fmaf(y, w1f[i0 + 1], b1f[i0 + 1]));
                hg[2 * r]     = (_Float16)TANHF(__builtin_fmaf(y, w1g[i0],     b1g[i0]));
                hg[2 * r + 1] = (_Float16)TANHF(__builtin_fmaf(y, w1g[i0 + 1], b1g[i0 + 1]));
            }
            df = __builtin_amdgcn_wmma_f32_16x16x32_f16(false, aF[c], false, hf,
                                                        (short)0, df, false, false);
            dg = __builtin_amdgcn_wmma_f32_16x16x32_f16(false, aG[c], false, hg,
                                                        (short)0, dg, false, false);
        }

        const float fv = TANHF(df[0] + b2f);   // drift
        const float gv = TANHF(dg[0] + b2g);   // diffusion

        const float dt = dt2[t][0], sq = dt2[t][1];
        const float z  = nbuf[w][cur][m];
        y = __builtin_fmaf(fv, dt, __builtin_fmaf(gv, sq * z, y)); // Euler-Maruyama
    }

    if (!hi) out[path] = __logf(y / yinit);
}

extern "C" void kernel_launch(void* const* d_in, const int* in_sizes, int n_in,
                              void* d_out, int out_size, void* d_ws, size_t ws_size,
                              hipStream_t stream)
{
    const float* y0    = (const float*)d_in[0];
    const float* times = (const float*)d_in[1];
    const float* noise = (const float*)d_in[2];
    const float* f1w   = (const float*)d_in[3];
    const float* f1b   = (const float*)d_in[4];
    const float* f2w   = (const float*)d_in[5];
    const float* f2b   = (const float*)d_in[6];
    const float* g1w   = (const float*)d_in[7];
    const float* g1b   = (const float*)d_in[8];
    const float* g2w   = (const float*)d_in[9];
    const float* g2b   = (const float*)d_in[10];

    const int N = in_sizes[0];   // 131072 paths
    const int T = in_sizes[1];   // 500 time points

    const int blocks = N / 128;  // 128 paths per 256-thread block (8 waves x 16)
    hipLaunchKernelGGL(deepsde_em_kernel, dim3(blocks), dim3(256), 0, stream,
                       y0, times, noise, f1w, f1b, f2w, f2b, g1w, g1b, g2w, g2b,
                       (float*)d_out, N, T);
}
